// GNN_72791105733293
// MI455X (gfx1250) — matrix operand order repaired and checked by reading, not compiled, not run on hardware
//
#include <hip/hip_runtime.h>
#include <hip/hip_bf16.h>

typedef __attribute__((ext_vector_type(16))) __bf16 v16bf;
typedef __attribute__((ext_vector_type(8)))  __bf16 v8bf;
typedef __attribute__((ext_vector_type(8)))  float  v8f;

#define NNODES 50000
#define NEDGES 800000
#define NGRAPH 128
#define DFEAT  256
#define NLAYER 5   // conv1 + 4 in convs

// ---------------------------------------------------------------------------
// WMMA GEMM: C[n,256] = A[n,KDIM] @ W[KDIM,256] (+bias)(+relu)
// bf16 inputs (converted during LDS staging), f32 accumulate via
// v_wmma_f32_16x16x32_bf16.
// Block tile 128x128, 8 waves, each wave computes a 16x128 strip:
//   8 WMMA accumulators per wave, one A-fragment load feeds 8 WMMAs.
// LDS tiles are double-buffered: one barrier per K-step, next tile's global
// loads + bf16 conversion overlap current tile's fragment loads + WMMAs.
// With a 128-wide N tile (grid.y == 2) the fp32 activations are only read
// twice per GEMM instead of 4x -- this pipeline is HBM-bound.
// ---------------------------------------------------------------------------
template<int KDIM>
__global__ __launch_bounds__(256)
void k_gemm_wmma(const float* __restrict__ A, const float* __restrict__ W,
                 const float* __restrict__ bias, float* __restrict__ C,
                 int n, int do_relu)
{
  __shared__ __bf16 As[2][128][40];   // 128 rows x 32 K (+8 pad), 80B row stride
  __shared__ __bf16 BsT[2][128][40];  // 128 cols x 32 K (transposed tile)

  const int tid  = threadIdx.x;
  const int wave = tid >> 5, lane = tid & 31;
  const int half = lane >> 4, nc = lane & 15;
  const int row0 = blockIdx.x * 128;
  const int col0 = blockIdx.y * 128;
  const int M = DFEAT;

  v8f acc[8] = {{}, {}, {}, {}, {}, {}, {}, {}};

  auto stage = [&](int kb, int buf) {
    // A tile: 128x32 f32 -> bf16, row-guarded
#pragma unroll
    for (int t = 0; t < 4; ++t) {
      int e  = t * 256 + tid;
      int r  = e >> 3;
      int c4 = (e & 7) * 4;
      int gr = row0 + r;
      float4 v = make_float4(0.f, 0.f, 0.f, 0.f);
      if (gr < n) v = *(const float4*)&A[(long)gr * KDIM + kb + c4];
      As[buf][r][c4 + 0] = (__bf16)v.x;
      As[buf][r][c4 + 1] = (__bf16)v.y;
      As[buf][r][c4 + 2] = (__bf16)v.z;
      As[buf][r][c4 + 3] = (__bf16)v.w;
    }
    // B tile: 32x128 f32 -> bf16, stored transposed (col-major in LDS)
#pragma unroll
    for (int t = 0; t < 4; ++t) {
      int e  = t * 256 + tid;
      int kr = e >> 5;
      int c4 = (e & 31) * 4;
      float4 v = *(const float4*)&W[(long)(kb + kr) * M + col0 + c4];
      BsT[buf][c4 + 0][kr] = (__bf16)v.x;
      BsT[buf][c4 + 1][kr] = (__bf16)v.y;
      BsT[buf][c4 + 2][kr] = (__bf16)v.z;
      BsT[buf][c4 + 3][kr] = (__bf16)v.w;
    }
  };

  stage(0, 0);
  constexpr int NT = KDIM / 32;
#pragma unroll
  for (int it = 0; it < NT; ++it) {
    __syncthreads();
    if (it + 1 < NT) stage((it + 1) * 32, (it + 1) & 1);  // prefetch next tile
    const int buf = it & 1;

    // A fragment: lane holds row M=(wave*16+nc), K = {8h..8h+7, 16+8h..+7}
    const int ar = wave * 16 + nc;
    v8bf alo = *(const v8bf*)&As[buf][ar][half * 8];
    v8bf ahi = *(const v8bf*)&As[buf][ar][16 + half * 8];
    v16bf a;
#pragma unroll
    for (int i = 0; i < 8; ++i) { a[i] = alo[i]; a[i + 8] = ahi[i]; }

    // 8 B fragments covering local cols 0..127: lane holds col N=s*16+nc,
    // K = {8h..8h+7, 16+8h..+7} (matches A fragment order)
#pragma unroll
    for (int s = 0; s < 8; ++s) {
      const int bc = s * 16 + nc;
      v8bf blo = *(const v8bf*)&BsT[buf][bc][half * 8];
      v8bf bhi = *(const v8bf*)&BsT[buf][bc][16 + half * 8];
      v16bf b;
#pragma unroll
      for (int i = 0; i < 8; ++i) { b[i] = blo[i]; b[i + 8] = bhi[i]; }
      acc[s] = __builtin_amdgcn_wmma_f32_16x16x32_bf16(false, a, false, b,
                                                       (short)0, acc[s], false, false);
    }
  }

  // epilogue: C/D layout -> M = r + 8*half, N = nc
  const int rbase = row0 + wave * 16;
#pragma unroll
  for (int r = 0; r < 8; ++r) {
    int gr = rbase + r + 8 * half;
    if (gr >= n) continue;
#pragma unroll
    for (int s = 0; s < 8; ++s) {
      int gc = col0 + s * 16 + nc;
      float v = acc[s][r];
      if (bias) v += bias[gc];
      if (do_relu) v = fmaxf(v, 0.f);
      C[(long)gr * M + gc] = v;
    }
  }
}

// ---------------------------------------------------------------------------
// Small helper kernels
// ---------------------------------------------------------------------------
__global__ void k_fill(float* p, long nelem, float v) {
  long i = (long)blockIdx.x * blockDim.x + threadIdx.x;
  if (i < nelem) p[i] = v;
}

__global__ void k_deg(const int* __restrict__ dst, float* __restrict__ deg, int ne) {
  int e = blockIdx.x * blockDim.x + threadIdx.x;
  if (e < ne) atomicAdd(&deg[dst[e]], 1.0f);
}

__global__ void k_rsqrt_inplace(float* d, int n) {
  int i = blockIdx.x * blockDim.x + threadIdx.x;
  if (i < n) d[i] = rsqrtf(d[i]);
}

// agg[i][f] = bias[f]
__global__ void k_rowbias(float* __restrict__ agg, const float* __restrict__ b, int n) {
  long i = (long)blockIdx.x * blockDim.x + threadIdx.x;
  if (i < (long)n * DFEAT) agg[i] = b[i & (DFEAT - 1)];
}

// agg[dst] += dis[src]*dis[dst] * m[src]   for E edges + N self loops.
// Two chunks per thread with a prefetch on the second source row so the
// random-gather latency overlaps the first chunk's atomics.
__global__ void k_scatter(const float* __restrict__ m, const float* __restrict__ dis,
                          const int* __restrict__ ei, float* __restrict__ agg) {
  const long half_total = (long)(NEDGES + NNODES) * 32;  // 64 chunks/edge, 2 per thread
  long gid = (long)blockIdx.x * blockDim.x + threadIdx.x;
  if (gid >= half_total) return;
  long id1 = gid;
  long id2 = gid + half_total;

  int e1 = (int)(id1 >> 6), c1 = (int)(id1 & 63);
  int e2 = (int)(id2 >> 6), c2 = (int)(id2 & 63);
  int s1, d1, s2, d2;
  if (e1 < NEDGES) { s1 = ei[e1]; d1 = ei[NEDGES + e1]; } else { s1 = d1 = e1 - NEDGES; }
  if (e2 < NEDGES) { s2 = ei[e2]; d2 = ei[NEDGES + e2]; } else { s2 = d2 = e2 - NEDGES; }

  __builtin_prefetch((const void*)&m[(long)s2 * DFEAT + c2 * 4], 0, 1);

  float nm1 = dis[s1] * dis[d1];
  float4 v1 = *(const float4*)&m[(long)s1 * DFEAT + c1 * 4];
  float* o1 = &agg[(long)d1 * DFEAT + c1 * 4];
  atomicAdd(o1 + 0, nm1 * v1.x);
  atomicAdd(o1 + 1, nm1 * v1.y);
  atomicAdd(o1 + 2, nm1 * v1.z);
  atomicAdd(o1 + 3, nm1 * v1.w);

  float nm2 = dis[s2] * dis[d2];
  float4 v2 = *(const float4*)&m[(long)s2 * DFEAT + c2 * 4];
  float* o2 = &agg[(long)d2 * DFEAT + c2 * 4];
  atomicAdd(o2 + 0, nm2 * v2.x);
  atomicAdd(o2 + 1, nm2 * v2.y);
  atomicAdd(o2 + 2, nm2 * v2.z);
  atomicAdd(o2 + 3, nm2 * v2.w);
}

// in-place relu + per-feature sum / sumsq partials (64 rows per block)
__global__ __launch_bounds__(DFEAT)
void k_relu_stats(float* __restrict__ h, float* __restrict__ stats, int n) {
  int f  = threadIdx.x;
  int r0 = blockIdx.x * 64;
  int r1 = r0 + 64; if (r1 > n) r1 = n;
  float s = 0.f, s2 = 0.f;
  for (int r = r0; r < r1; ++r) {
    float v = h[(long)r * DFEAT + f];
    v = fmaxf(v, 0.f);
    h[(long)r * DFEAT + f] = v;
    s += v; s2 += v * v;
  }
  atomicAdd(&stats[f], s);
  atomicAdd(&stats[DFEAT + f], s2);
}

__global__ void k_bn_apply(float* __restrict__ h, const float* __restrict__ stats,
                           const float* __restrict__ g, const float* __restrict__ bb,
                           int n) {
  long i = (long)blockIdx.x * blockDim.x + threadIdx.x;
  if (i >= (long)n * DFEAT) return;
  int f = (int)(i & (DFEAT - 1));
  float inv_n = 1.0f / (float)n;
  float mu  = stats[f] * inv_n;
  float var = stats[DFEAT + f] * inv_n - mu * mu;
  h[i] = g[f] * (h[i] - mu) * rsqrtf(var + 1e-5f) + bb[f];
}

__global__ void k_pool(const float* __restrict__ h, const int* __restrict__ batch,
                       float* __restrict__ sums, float* __restrict__ cnt) {
  long gid = (long)blockIdx.x * blockDim.x + threadIdx.x;
  const long total = (long)NNODES * 64;
  if (gid >= total) return;
  int i = (int)(gid >> 6);
  int c = (int)(gid & 63);
  int g = batch[i];
  float4 v = *(const float4*)&h[(long)i * DFEAT + c * 4];
  float* op = &sums[(long)g * DFEAT + c * 4];
  atomicAdd(op + 0, v.x);
  atomicAdd(op + 1, v.y);
  atomicAdd(op + 2, v.z);
  atomicAdd(op + 3, v.w);
  if (c == 0) atomicAdd(&cnt[g], 1.0f);
}

__global__ void k_pool_div(float* __restrict__ sums, const float* __restrict__ cnt) {
  int i = blockIdx.x * blockDim.x + threadIdx.x;
  if (i >= NGRAPH * DFEAT) return;
  int g = i >> 8;  // DFEAT == 256
  sums[i] /= fmaxf(cnt[g], 1.0f);
}

// small dense: out[g][j] = (relu)(sum_k in[g][k % Kmod] * W[k*Mo+j] + b[j])
// Kmod < Kin implements concat(pooled, pooled).
__global__ __launch_bounds__(512)
void k_dense(const float* __restrict__ in, const float* __restrict__ W,
             const float* __restrict__ b, float* __restrict__ out,
             int Kin, int Kmod, int Mo, int do_relu) {
  __shared__ float srow[512];
  int g = blockIdx.x;
  for (int k = threadIdx.x; k < Kin; k += blockDim.x)
    srow[k] = in[(long)g * Kmod + (k % Kmod)];
  __syncthreads();
  for (int j = threadIdx.x; j < Mo; j += blockDim.x) {
    float acc = b[j];
    for (int k = 0; k < Kin; ++k) acc += srow[k] * W[(long)k * Mo + j];
    out[(long)g * Mo + j] = do_relu ? fmaxf(acc, 0.f) : acc;
  }
}

// ---------------------------------------------------------------------------
// Host-side launch
// ---------------------------------------------------------------------------
extern "C" void kernel_launch(void* const* d_in, const int* in_sizes, int n_in,
                              void* d_out, int out_size, void* d_ws, size_t ws_size,
                              hipStream_t stream) {
  (void)in_sizes; (void)n_in; (void)out_size; (void)ws_size;

  const float* x       = (const float*)d_in[0];
  const int*   ei      = (const int*)d_in[1];   // [2, E]: src row then dst row
  const int*   batch   = (const int*)d_in[2];
  const float* fl_W    = (const float*)d_in[3];
  const float* fl_b    = (const float*)d_in[4];
  const float* conv1_W = (const float*)d_in[5];
  const float* conv1_b = (const float*)d_in[6];
  const float* tr1_W   = (const float*)d_in[7];
  const float* tr1_b   = (const float*)d_in[8];
  const float* bn1_g   = (const float*)d_in[9];
  const float* bn1_b   = (const float*)d_in[10];
  const float* convs_W = (const float*)d_in[11];
  const float* convs_b = (const float*)d_in[12];
  const float* trs_W   = (const float*)d_in[13];
  const float* trs_b   = (const float*)d_in[14];
  const float* bns_g   = (const float*)d_in[15];
  const float* bns_b   = (const float*)d_in[16];
  const float* l1_W    = (const float*)d_in[17];
  const float* l1_b    = (const float*)d_in[18];
  const float* l2_W    = (const float*)d_in[19];
  const float* l2_b    = (const float*)d_in[20];
  const float* l3_W    = (const float*)d_in[21];
  const float* l3_b    = (const float*)d_in[22];

  // workspace carving (floats)
  float* ws = (float*)d_ws;
  const size_t NB = (size_t)NNODES * DFEAT;
  float* h     = ws;                    // [N,256] activations
  float* m     = ws + NB;               // [N,256] h @ convW
  float* agg   = ws + 2 * NB;           // [N,256] scatter result
  float* dis   = ws + 3 * NB;           // [N] deg -> rsqrt(deg)
  float* stats = dis + NNODES;          // [512] sum | sumsq
  float* sums  = stats + 512;           // [G,256] pooled
  float* cnt   = sums + (size_t)NGRAPH * DFEAT;  // [G]
  float* z1    = cnt + NGRAPH;          // [G,512]
  float* z2    = z1 + (size_t)NGRAPH * 512;      // [G,256]

  const dim3 gemm_grid((NNODES + 127) / 128, DFEAT / 128);

  // degrees (self loop => init 1.0), then dis = rsqrt(deg)
  k_fill<<<(NNODES + 255) / 256, 256, 0, stream>>>(dis, NNODES, 1.0f);
  k_deg<<<(NEDGES + 255) / 256, 256, 0, stream>>>(ei + NEDGES, dis, NEDGES);
  k_rsqrt_inplace<<<(NNODES + 255) / 256, 256, 0, stream>>>(dis, NNODES);

  // h = x @ fl_W + fl_b
  k_gemm_wmma<128><<<gemm_grid, 256, 0, stream>>>(x, fl_W, fl_b, h, NNODES, 0);

  const long scat_blocks = ((long)(NEDGES + NNODES) * 32 + 255) / 256;
  const long elem_blocks = ((long)NNODES * DFEAT + 255) / 256;

  for (int layer = 0; layer < NLAYER; ++layer) {
    const float* cW = (layer == 0) ? conv1_W : convs_W + (size_t)(layer - 1) * DFEAT * DFEAT;
    const float* cB = (layer == 0) ? conv1_b : convs_b + (size_t)(layer - 1) * DFEAT;
    const float* tW = (layer == 0) ? tr1_W   : trs_W  + (size_t)(layer - 1) * DFEAT * DFEAT;
    const float* tB = (layer == 0) ? tr1_b   : trs_b  + (size_t)(layer - 1) * DFEAT;
    const float* bG = (layer == 0) ? bn1_g   : bns_g  + (size_t)(layer - 1) * DFEAT;
    const float* bB = (layer == 0) ? bn1_b   : bns_b  + (size_t)(layer - 1) * DFEAT;

    // m = h @ convW
    k_gemm_wmma<256><<<gemm_grid, 256, 0, stream>>>(h, cW, nullptr, m, NNODES, 0);
    // agg = bias, then scatter-add norm * m[src]
    k_rowbias<<<elem_blocks, 256, 0, stream>>>(agg, cB, NNODES);
    k_scatter<<<scat_blocks, 256, 0, stream>>>(m, dis, ei, agg);
    // relu + BN (train-mode biased stats)
    k_fill<<<2, 256, 0, stream>>>(stats, 512, 0.0f);
    k_relu_stats<<<(NNODES + 63) / 64, DFEAT, 0, stream>>>(agg, stats, NNODES);
    k_bn_apply<<<elem_blocks, 256, 0, stream>>>(agg, stats, bG, bB, NNODES);
    // h = relu(agg @ trW + trb)
    k_gemm_wmma<256><<<gemm_grid, 256, 0, stream>>>(agg, tW, tB, h, NNODES, 1);
  }

  // global mean pool
  k_fill<<<(NGRAPH * DFEAT + 255) / 256, 256, 0, stream>>>(sums, NGRAPH * DFEAT, 0.0f);
  k_fill<<<1, 256, 0, stream>>>(cnt, NGRAPH, 0.0f);
  k_pool<<<((long)NNODES * 64 + 255) / 256, 256, 0, stream>>>(h, batch, sums, cnt);
  k_pool_div<<<(NGRAPH * DFEAT + 255) / 256, 256, 0, stream>>>(sums, cnt);

  // MLP head: z = concat(pooled, pooled) handled via Kmod indexing
  k_dense<<<NGRAPH, 512, 0, stream>>>(sums, l1_W, l1_b, z1, 512, 256, 512, 1);
  k_dense<<<NGRAPH, 512, 0, stream>>>(z1, l2_W, l2_b, z2, 512, 512, 256, 1);
  k_dense<<<NGRAPH, 512, 0, stream>>>(z2, l3_W, l3_b, (float*)d_out, 256, 256, 10, 0);
}
